// TransformerBlock_82188494176807
// MI455X (gfx1250) — compile-verified, matrix operand order checked by reading, not run on hardware
//
#include <hip/hip_runtime.h>
#include <hip/hip_bf16.h>
#include <math.h>

typedef __attribute__((ext_vector_type(16))) __bf16 v16bf;
typedef __attribute__((ext_vector_type(8)))  float  v8f;
typedef __attribute__((ext_vector_type(4)))  int    v4i;

#define EMBD   1024
#define NHEAD  16
#define HDIM   64
#define FFDIM  4096
#define SEQ    2048
#define NBATCH 4
#define TOK    (NBATCH * SEQ)   // 8192 tokens

// ---------------------------------------------------- async LDS staging -----
#if __has_builtin(__builtin_amdgcn_global_load_async_to_lds_b128) && \
    __has_builtin(__builtin_amdgcn_s_wait_asynccnt)
#define ASYNC_LDS 1
typedef __attribute__((address_space(1))) v4i gas_v4i;
typedef __attribute__((address_space(3))) v4i las_v4i;
#else
#define ASYNC_LDS 0
#endif

__device__ __forceinline__ void copy16_to_lds(const void* g, void* l) {
#if ASYNC_LDS
  __builtin_amdgcn_global_load_async_to_lds_b128((gas_v4i*)g, (las_v4i*)l, 0, 0);
#else
  *(uint4*)l = *(const uint4*)g;
#endif
}
__device__ __forceinline__ void wait_async_lds() {
#if ASYNC_LDS
  __builtin_amdgcn_s_wait_asynccnt(0);
#endif
}

// ------------------------------------------ 16-lane butterfly reductions ----
// xor-by-m (m in 1,2,4,8) within each 16-lane row via v_permlane16_b32
// (VALU, co-executes with WMMA) instead of ds_bpermute LDS round-trips.
#if __has_builtin(__builtin_amdgcn_permlane16)
#define HAVE_PL16 1
#else
#define HAVE_PL16 0
#endif

__device__ __forceinline__ float pl16(float v, unsigned lo, unsigned hi, int m) {
#if HAVE_PL16
  unsigned u = __float_as_uint(v);
  return __uint_as_float(__builtin_amdgcn_permlane16(u, u, lo, hi, false, false));
#else
  return __shfl_xor(v, m, 32);
#endif
}
__device__ __forceinline__ float half16_max(float v) {
  v = fmaxf(v, pl16(v, 0x67452301u, 0xEFCDAB89u, 1));  // ^1
  v = fmaxf(v, pl16(v, 0x54761032u, 0xDCFE98BAu, 2));  // ^2
  v = fmaxf(v, pl16(v, 0x32107654u, 0xBA98FEDCu, 4));  // ^4
  v = fmaxf(v, pl16(v, 0xFEDCBA98u, 0x76543210u, 8));  // ^8
  return v;
}
__device__ __forceinline__ float half16_sum(float v) {
  v += pl16(v, 0x67452301u, 0xEFCDAB89u, 1);
  v += pl16(v, 0x54761032u, 0xDCFE98BAu, 2);
  v += pl16(v, 0x32107654u, 0xBA98FEDCu, 4);
  v += pl16(v, 0xFEDCBA98u, 0x76543210u, 8);
  return v;
}

__device__ __forceinline__ float gelu_tanh_f(float v) {
  const float c = 0.7978845608028654f;  // sqrt(2/pi)
  return 0.5f * v * (1.0f + tanhf(c * (v + 0.044715f * v * v * v)));
}

// Load a 16x32 bf16 fragment in the CDNA5 WMMA A/B register layout:
// lanes 0-15: row = lane,    K = {0..7, 16..23}
// lanes 16-31: row = lane-16, K = {8..15, 24..31}
__device__ __forceinline__ v16bf load_frag_k32(const __bf16* base, int ld) {
  const int lane = threadIdx.x & 31;
  const int row  = lane & 15;
  const int koff = (lane < 16) ? 0 : 8;
  const __bf16* p = base + row * ld + koff;
  v16bf f;
#pragma unroll
  for (int i = 0; i < 8; ++i) f[i] = p[i];
#pragma unroll
  for (int i = 0; i < 8; ++i) f[i + 8] = p[16 + i];
  return f;
}

// ------------------------------------------------------- weight transpose ---
__global__ void transpose_to_bf16(const float* __restrict__ W,
                                  __bf16* __restrict__ Wt, int K, int N) {
  long long idx = (long long)blockIdx.x * blockDim.x + threadIdx.x;
  long long total = (long long)K * N;
  if (idx >= total) return;
  int n = (int)(idx / K);
  int k = (int)(idx % K);
  Wt[(size_t)n * K + k] = (__bf16)W[(size_t)k * N + n];
}

// ------------------------------------------------------------- layernorm ---
__global__ __launch_bounds__(256)
void layernorm_to_bf16(const float* __restrict__ X, const float* __restrict__ g,
                       const float* __restrict__ b, __bf16* __restrict__ H) {
  const int row = blockIdx.x;
  const float* x = X + (size_t)row * EMBD;
  float s = 0.f, s2 = 0.f;
  for (int i = threadIdx.x; i < EMBD; i += 256) {
    float v = x[i]; s += v; s2 += v * v;
  }
#pragma unroll
  for (int m = 1; m < 32; m <<= 1) {
    s  += __shfl_xor(s,  m, 32);
    s2 += __shfl_xor(s2, m, 32);
  }
  __shared__ float ws[8], ws2[8];
  const int wave = threadIdx.x >> 5;
  if ((threadIdx.x & 31) == 0) { ws[wave] = s; ws2[wave] = s2; }
  __syncthreads();
  s = 0.f; s2 = 0.f;
#pragma unroll
  for (int i = 0; i < 8; ++i) { s += ws[i]; s2 += ws2[i]; }
  const float mean = s * (1.0f / EMBD);
  const float var  = (s2 - (float)EMBD * mean * mean) * (1.0f / (EMBD - 1));
  const float rstd = rsqrtf(var + 1e-5f);
  __bf16* h = H + (size_t)row * EMBD;
  for (int i = threadIdx.x; i < EMBD; i += 256)
    h[i] = (__bf16)((x[i] - mean) * rstd * g[i] + b[i]);
}

// ---------------------------------------------------------- WMMA GEMM -------
// C[M,N] = A[M,K](bf16,row) x Bt[N,K](bf16,row)^T  (+bias)(gelu)(+res)
// 256 thr = 8 waves; C tile 128x64; wave -> 32x32; K-step 64.
// LDS double-buffered, staged with async-to-LDS loads (ASYNCcnt).
template <bool BIAS, bool GELU, bool RES, bool OUTBF>
__global__ __launch_bounds__(256)
void gemm_bf16_wmma(const __bf16* __restrict__ A, const __bf16* __restrict__ Bt,
                    const float* __restrict__ bias, const float* __restrict__ res,
                    float* __restrict__ Cf, __bf16* __restrict__ Cb,
                    int M, int N, int K) {
  __shared__ __align__(16) __bf16 As[2][128 * 64];  // 2 x 16KB
  __shared__ __align__(16) __bf16 Bs[2][64 * 64];   // 2 x  8KB
  const int tid  = threadIdx.x;
  const int lane = tid & 31;
  const int wave = tid >> 5;
  const int mbase = blockIdx.y * 128;
  const int nbase = blockIdx.x * 64;
  const int rm = (wave & 3) * 32;
  const int cn = (wave >> 2) * 32;

  auto stage = [&](int k0, int buf) {
#pragma unroll
    for (int i = 0; i < 4; ++i) {           // A: 128x64 bf16 = 1024 x 16B
      const int t = tid + 256 * i;
      const int r = t >> 3, c = (t & 7) * 8;
      copy16_to_lds(&A[(size_t)(mbase + r) * K + k0 + c], &As[buf][r * 64 + c]);
    }
#pragma unroll
    for (int i = 0; i < 2; ++i) {           // B: 64x64 bf16 = 512 x 16B
      const int t = tid + 256 * i;
      const int r = t >> 3, c = (t & 7) * 8;
      copy16_to_lds(&Bt[(size_t)(nbase + r) * K + k0 + c], &Bs[buf][r * 64 + c]);
    }
  };

  v8f acc[2][2];
#pragma unroll
  for (int m2 = 0; m2 < 2; ++m2)
#pragma unroll
    for (int n2 = 0; n2 < 2; ++n2) {
      v8f z = {0.f, 0.f, 0.f, 0.f, 0.f, 0.f, 0.f, 0.f};
      acc[m2][n2] = z;
    }

  stage(0, 0);
  int cur = 0;
  for (int k0 = 0; k0 < K; k0 += 64) {
    wait_async_lds();      // this wave's staged tiles landed
    __syncthreads();       // ... and everyone else's too
    if (k0 + 64 < K) stage(k0 + 64, cur ^ 1);  // prefetch next K-step

    v16bf a[2][2], b[2][2];
#pragma unroll
    for (int m2 = 0; m2 < 2; ++m2)
#pragma unroll
      for (int kh = 0; kh < 2; ++kh)
        a[m2][kh] = load_frag_k32(&As[cur][(rm + m2 * 16) * 64 + kh * 32], 64);
#pragma unroll
    for (int n2 = 0; n2 < 2; ++n2)
#pragma unroll
      for (int kh = 0; kh < 2; ++kh)
        b[n2][kh] = load_frag_k32(&Bs[cur][(cn + n2 * 16) * 64 + kh * 32], 64);
#pragma unroll
    for (int m2 = 0; m2 < 2; ++m2)
#pragma unroll
      for (int n2 = 0; n2 < 2; ++n2) {
        acc[m2][n2] = __builtin_amdgcn_wmma_f32_16x16x32_bf16(
            false, a[m2][0], false, b[n2][0], (short)0, acc[m2][n2], false, false);
        acc[m2][n2] = __builtin_amdgcn_wmma_f32_16x16x32_bf16(
            false, a[m2][1], false, b[n2][1], (short)0, acc[m2][n2], false, false);
      }
    cur ^= 1;
  }

  const int hi  = lane >> 4;
  const int col = lane & 15;
#pragma unroll
  for (int m2 = 0; m2 < 2; ++m2)
#pragma unroll
    for (int n2 = 0; n2 < 2; ++n2)
#pragma unroll
      for (int j = 0; j < 8; ++j) {
        const size_t grow = (size_t)(mbase + rm + m2 * 16 + j + 8 * hi);
        const int gcol = nbase + cn + n2 * 16 + col;
        float v = acc[m2][n2][j];
        if (BIAS) v += bias[gcol];
        if (GELU) v = gelu_tanh_f(v);
        if (RES)  v += res[grow * N + gcol];
        if (OUTBF) Cb[grow * N + gcol] = (__bf16)v;
        else       Cf[grow * N + gcol] = v;
      }
}

// ------------------------------------------------- flash attention (WMMA) ---
// one wave per (batch, head, 32-query tile); causal, online softmax.
// 32 queries/wave: K and V fragments are loaded once per key block and
// reused by both 16-row q-tiles (16 WMMAs per 32-key block).
template <bool MASK>
__device__ __forceinline__ void attn_kblock32(
    int kb, int qb, const __bf16* Kp, const __bf16* Vp,
    const v16bf (&aq)[2][2],
    float (&mx)[2][8], float (&l)[2][8], v8f (&acc)[2][4], __bf16* Plds) {
  const int lane = threadIdx.x & 31;
  const int hi   = lane >> 4;
  const int col  = lane & 15;
  const float scale = 0.125f;  // 1/sqrt(64)

  v16bf bk[2][2];
#pragma unroll
  for (int u = 0; u < 2; ++u)
#pragma unroll
    for (int kh = 0; kh < 2; ++kh)
      bk[u][kh] = load_frag_k32(Kp + (size_t)(kb + u * 16) * EMBD + kh * 32, EMBD);

  v8f s[2][2];
#pragma unroll
  for (int t = 0; t < 2; ++t)
#pragma unroll
    for (int u = 0; u < 2; ++u) {
      v8f z = {0.f, 0.f, 0.f, 0.f, 0.f, 0.f, 0.f, 0.f};
      z = __builtin_amdgcn_wmma_f32_16x16x32_bf16(false, aq[t][0], false, bk[u][0],
                                                  (short)0, z, false, false);
      z = __builtin_amdgcn_wmma_f32_16x16x32_bf16(false, aq[t][1], false, bk[u][1],
                                                  (short)0, z, false, false);
      s[t][u] = z;
    }

#pragma unroll
  for (int t = 0; t < 2; ++t)
#pragma unroll
    for (int j = 0; j < 8; ++j) {
      float v0 = s[t][0][j] * scale;
      float v1 = s[t][1][j] * scale;
      if (MASK) {
        const int q = qb + t * 16 + j + 8 * hi;
        v0 = (kb + col      > q) ? -3.0e38f : v0;
        v1 = (kb + 16 + col > q) ? -3.0e38f : v1;
      }
      const float bm   = half16_max(fmaxf(v0, v1));
      const float mnew = fmaxf(mx[t][j], bm);
      const float corr = __expf(mx[t][j] - mnew);
      const float p0 = __expf(v0 - mnew);
      const float p1 = __expf(v1 - mnew);
      l[t][j] = l[t][j] * corr + half16_sum(p0 + p1);
      mx[t][j] = mnew;
      acc[t][0][j] *= corr; acc[t][1][j] *= corr;
      acc[t][2][j] *= corr; acc[t][3][j] *= corr;
      Plds[t * 512 + (j + 8 * hi) * 32 + col]      = (__bf16)p0;
      Plds[t * 512 + (j + 8 * hi) * 32 + 16 + col] = (__bf16)p1;
    }
  asm volatile("s_wait_dscnt 0" ::: "memory");  // P visible across lanes
  const v16bf ap0 = load_frag_k32(Plds, 32);
  const v16bf ap1 = load_frag_k32(Plds + 512, 32);

#pragma unroll
  for (int f = 0; f < 4; ++f) {  // O += P V, 4 d-tiles; bv shared by both tiles
    const __bf16* vp =
        Vp + (size_t)(kb + ((lane < 16) ? 0 : 8)) * EMBD + f * 16 + col;
    v16bf bv;
#pragma unroll
    for (int i = 0; i < 8; ++i) bv[i] = vp[(size_t)i * EMBD];
#pragma unroll
    for (int i = 0; i < 8; ++i) bv[8 + i] = vp[(size_t)(16 + i) * EMBD];
    acc[0][f] = __builtin_amdgcn_wmma_f32_16x16x32_bf16(false, ap0, false, bv,
                                                        (short)0, acc[0][f], false, false);
    acc[1][f] = __builtin_amdgcn_wmma_f32_16x16x32_bf16(false, ap1, false, bv,
                                                        (short)0, acc[1][f], false, false);
  }
  asm volatile("s_wait_dscnt 0" ::: "memory");  // P reads done before reuse
}

__global__ __launch_bounds__(32)
void flash_attn_wmma(const __bf16* __restrict__ Qb, const __bf16* __restrict__ Kb,
                     const __bf16* __restrict__ Vb, __bf16* __restrict__ Ctx) {
  __shared__ __align__(16) __bf16 Plds[2 * 16 * 32];
  const int lane  = threadIdx.x & 31;
  const int hi    = lane >> 4;
  const int col   = lane & 15;
  const int qb    = blockIdx.x * 32;  // 32-aligned query tile
  const int head  = blockIdx.y;
  const int batch = blockIdx.z;
  const size_t base = (size_t)batch * SEQ * EMBD + head * HDIM;
  const __bf16* Qp = Qb + base;
  const __bf16* Kp = Kb + base;
  const __bf16* Vp = Vb + base;

  v16bf aq[2][2];
#pragma unroll
  for (int t = 0; t < 2; ++t)
#pragma unroll
    for (int kh = 0; kh < 2; ++kh)
      aq[t][kh] = load_frag_k32(Qp + (size_t)(qb + t * 16) * EMBD + kh * 32, EMBD);

  float mx[2][8], l[2][8];
  v8f acc[2][4];
#pragma unroll
  for (int t = 0; t < 2; ++t)
#pragma unroll
    for (int j = 0; j < 8; ++j) { mx[t][j] = -3.0e38f; l[t][j] = 0.f; }
#pragma unroll
  for (int t = 0; t < 2; ++t)
#pragma unroll
    for (int f = 0; f < 4; ++f) {
      v8f z = {0.f, 0.f, 0.f, 0.f, 0.f, 0.f, 0.f, 0.f};
      acc[t][f] = z;
    }

  int kb = 0;
  for (; kb + 31 <= qb; kb += 32)   // fully unmasked blocks
    attn_kblock32<false>(kb, qb, Kp, Vp, aq, mx, l, acc, Plds);
  for (; kb <= qb + 31; kb += 32)   // diagonal (masked) block
    attn_kblock32<true>(kb, qb, Kp, Vp, aq, mx, l, acc, Plds);

#pragma unroll
  for (int t = 0; t < 2; ++t)
#pragma unroll
    for (int f = 0; f < 4; ++f)
#pragma unroll
      for (int j = 0; j < 8; ++j) {
        const size_t row = (size_t)batch * SEQ + qb + t * 16 + j + 8 * hi;
        Ctx[row * EMBD + head * HDIM + f * 16 + col] =
            (__bf16)(acc[t][f][j] / l[t][j]);
      }
}

// --------------------------------------------------------------- launcher ---
extern "C" void kernel_launch(void* const* d_in, const int* in_sizes, int n_in,
                              void* d_out, int out_size, void* d_ws, size_t ws_size,
                              hipStream_t stream) {
  const float* x   = (const float*)d_in[0];
  const float* Wq  = (const float*)d_in[1];
  const float* Wk  = (const float*)d_in[2];
  const float* Wv  = (const float*)d_in[3];
  const float* Wo  = (const float*)d_in[4];
  const float* bo  = (const float*)d_in[5];
  const float* g1  = (const float*)d_in[6];
  const float* b1n = (const float*)d_in[7];
  const float* g2  = (const float*)d_in[8];
  const float* b2n = (const float*)d_in[9];
  const float* W1  = (const float*)d_in[10];
  const float* b1  = (const float*)d_in[11];
  const float* W2  = (const float*)d_in[12];
  const float* b2  = (const float*)d_in[13];
  float* out = (float*)d_out;

  char* cur = (char*)d_ws;
  auto carve = [&](size_t bytes) {
    void* r = (void*)cur;
    cur += (bytes + 255) & ~(size_t)255;
    return r;
  };
  __bf16* h_bf = (__bf16*)carve((size_t)TOK * EMBD * 2);
  __bf16* WqT  = (__bf16*)carve((size_t)EMBD * EMBD * 2);
  __bf16* WkT  = (__bf16*)carve((size_t)EMBD * EMBD * 2);
  __bf16* WvT  = (__bf16*)carve((size_t)EMBD * EMBD * 2);
  __bf16* WoT  = (__bf16*)carve((size_t)EMBD * EMBD * 2);
  __bf16* W1T  = (__bf16*)carve((size_t)FFDIM * EMBD * 2);
  __bf16* W2T  = (__bf16*)carve((size_t)EMBD * FFDIM * 2);
  __bf16* Qb   = (__bf16*)carve((size_t)TOK * EMBD * 2);
  __bf16* Kb   = (__bf16*)carve((size_t)TOK * EMBD * 2);
  __bf16* Vb   = (__bf16*)carve((size_t)TOK * EMBD * 2);
  __bf16* ctx  = (__bf16*)carve((size_t)TOK * EMBD * 2);
  __bf16* h2   = (__bf16*)carve((size_t)TOK * EMBD * 2);
  __bf16* act  = (__bf16*)carve((size_t)TOK * FFDIM * 2);

  // 1) weight transpose+convert to bf16
  {
    int t1 = EMBD * EMBD, t2 = EMBD * FFDIM;
    transpose_to_bf16<<<(t1 + 255) / 256, 256, 0, stream>>>(Wq, WqT, EMBD, EMBD);
    transpose_to_bf16<<<(t1 + 255) / 256, 256, 0, stream>>>(Wk, WkT, EMBD, EMBD);
    transpose_to_bf16<<<(t1 + 255) / 256, 256, 0, stream>>>(Wv, WvT, EMBD, EMBD);
    transpose_to_bf16<<<(t1 + 255) / 256, 256, 0, stream>>>(Wo, WoT, EMBD, EMBD);
    transpose_to_bf16<<<(t2 + 255) / 256, 256, 0, stream>>>(W1, W1T, EMBD, FFDIM);
    transpose_to_bf16<<<(t2 + 255) / 256, 256, 0, stream>>>(W2, W2T, FFDIM, EMBD);
  }

  // 2) LN1 -> h (bf16)
  layernorm_to_bf16<<<TOK, 256, 0, stream>>>(x, g1, b1n, h_bf);

  // 3) QKV projections (token-major bf16)
  dim3 gE(EMBD / 64, TOK / 128);
  gemm_bf16_wmma<false, false, false, true><<<gE, 256, 0, stream>>>(
      h_bf, WqT, nullptr, nullptr, nullptr, Qb, TOK, EMBD, EMBD);
  gemm_bf16_wmma<false, false, false, true><<<gE, 256, 0, stream>>>(
      h_bf, WkT, nullptr, nullptr, nullptr, Kb, TOK, EMBD, EMBD);
  gemm_bf16_wmma<false, false, false, true><<<gE, 256, 0, stream>>>(
      h_bf, WvT, nullptr, nullptr, nullptr, Vb, TOK, EMBD, EMBD);

  // 4) causal flash attention (32 queries per wave)
  flash_attn_wmma<<<dim3(SEQ / 32, NHEAD, NBATCH), 32, 0, stream>>>(Qb, Kb, Vb, ctx);

  // 5) output projection + bias + residual -> d_out (fp32)
  gemm_bf16_wmma<true, false, true, false><<<gE, 256, 0, stream>>>(
      ctx, WoT, bo, x, out, nullptr, TOK, EMBD, EMBD);

  // 6) LN2 on d_out -> h2 (bf16)
  layernorm_to_bf16<<<TOK, 256, 0, stream>>>(out, g2, b2n, h2);

  // 7) MLP up + GELU -> act (bf16)
  dim3 gF(FFDIM / 64, TOK / 128);
  gemm_bf16_wmma<true, true, false, true><<<gF, 256, 0, stream>>>(
      h2, W1T, b1, nullptr, nullptr, act, TOK, FFDIM, EMBD);

  // 8) MLP down + bias + residual (in-place on d_out)
  gemm_bf16_wmma<true, false, true, false><<<gE, 256, 0, stream>>>(
      act, W2T, b2, out, out, nullptr, TOK, EMBD, FFDIM);

  (void)in_sizes; (void)n_in; (void)out_size; (void)ws_size;
}
